// ConditionalRandomField_25580825215210
// MI455X (gfx1250) — matrix operand order repaired and checked by reading, not compiled, not run on hardware
//
#include <hip/hip_runtime.h>

typedef __attribute__((ext_vector_type(16))) _Float16 v16h;
typedef __attribute__((ext_vector_type(8)))  _Float16 v8h;
typedef __attribute__((ext_vector_type(8)))  float    v8f;

constexpr int Bb = 128, Tt = 512, Cc = 128;

// ---------------------------------------------------------------------------
// Kernel 1: real path score (emission + transition) per batch, via one-hot
// label recovery. One block per batch, 256 threads (8 waves).
// ---------------------------------------------------------------------------
__global__ __launch_bounds__(256, 1)
void crf_score_kernel(const float* __restrict__ y_true,
                      const float* __restrict__ y_pred,
                      const float* __restrict__ mask,
                      const float* __restrict__ trans,
                      float* __restrict__ real_out)
{
    __shared__ int   s_lab[Tt];
    __shared__ float s_red[256];

    const int b    = blockIdx.x;
    const int tid  = threadIdx.x;
    const int wave = tid >> 5;
    const int lane = tid & 31;

    const float* yt = y_true + (size_t)b * Tt * Cc;
    const float* yp = y_pred + (size_t)b * Tt * Cc;
    const float* mk = mask   + (size_t)b * Tt;

    // Phase A: recover labels. One-hot => exactly one lane writes per t.
    for (int t = wave; t < Tt; t += 8) {
        const float* row = yt + (size_t)t * Cc;
#pragma unroll
        for (int ch = 0; ch < 4; ++ch) {
            int c = ch * 32 + lane;
            if (row[c] > 0.5f) s_lab[t] = c;
        }
    }
    __syncthreads();

    // Phase B: emission (mask^2 factor, faithful to reference) + transition.
    float acc = 0.f;
    for (int t = tid; t < Tt; t += 256) {
        float m = mk[t];
        acc += yp[(size_t)t * Cc + s_lab[t]] * m * m;
    }
    for (int t = tid; t < Tt - 1; t += 256) {
        acc += trans[s_lab[t] * Cc + s_lab[t + 1]] * mk[t] * mk[t + 1];
    }
    s_red[tid] = acc;
    __syncthreads();
    for (int off = 128; off > 0; off >>= 1) {
        if (tid < off) s_red[tid] += s_red[tid + off];
        __syncthreads();
    }
    if (tid == 0) real_out[b] = s_red[0];
}

// ---------------------------------------------------------------------------
// Kernel 2: forward recursion as per-step GEMM  S = exp(state-m) @ exp(trans)
// using v_wmma_f32_16x16x32_f16 with hi/lo f16 error splitting.
// 8 blocks (16 batches each) x 256 threads (8 waves, one 16-col tile each).
// ---------------------------------------------------------------------------
__global__ __launch_bounds__(256, 1)
void crf_forward_kernel(const float* __restrict__ y_pred,
                        const float* __restrict__ mask,
                        const float* __restrict__ trans,
                        const float* __restrict__ real_in,
                        float* __restrict__ out)
{
    __shared__ float s_state[16][Cc];     // 8 KB: current alpha (16 batches x C)
    __shared__ v16h  s_Ahi[4][32];        // 4 KB: A operand (hi), WMMA lane layout
    __shared__ v16h  s_Alo[4][32];        // 4 KB: A operand (lo)
    __shared__ float s_pmax[16][16];
    __shared__ float s_m[16];

    const int tid   = threadIdx.x;
    const int wave  = tid >> 5;
    const int lane  = tid & 31;
    const int rbase = blockIdx.x * 16;

    // --- Build B operands once: E = exp(trans) split hi/lo f16, kept in VGPRs.
    // B k order used here (contiguous, 32x16): lanes 0-15 hold K=0..15 (N=lane), lanes 16-31
    // hold K=16..31 (N=lane-16); element e of v16h is K = 16*(lane>=16)+e.
    v16h Ehi[4], Elo[4];
    {
        const int n = wave * 16 + (lane & 15);
#pragma unroll
        for (int c = 0; c < 4; ++c) {
            const int kb = 32 * c + 16 * (lane >> 4);
#pragma unroll
            for (int e = 0; e < 16; ++e) {
                float ev = __expf(trans[(kb + e) * Cc + n]);
                _Float16 h = (_Float16)ev;
                Ehi[c][e] = h;
                Elo[c][e] = (_Float16)(ev - (float)h);
            }
        }
    }

    // --- init state = y_pred[:,0] * mask[:,0]
    for (int i = tid; i < 16 * Cc; i += 256) {
        int r = i >> 7, c = i & (Cc - 1);
        s_state[r][c] = y_pred[((size_t)(rbase + r) * Tt) * Cc + c]
                      * mask[(size_t)(rbase + r) * Tt];
    }
    __syncthreads();

    // A-swizzle constants for the exp/store phase (thread -> 8 consecutive K).
    // A k order used here (contiguous, 16x32): lanes 0-15: e=0..15 -> K=0..15;
    // lanes 16-31: e=0..15 -> K=16..31.
    const int rA      = tid >> 4;
    const int kbA     = (tid & 15) * 8;
    const int kchunkA = kbA >> 5;
    const int kkA     = kbA & 31;
    const int laneA   = rA + 16 * ((kkA >> 4) & 1);
    const int halfA   = (kkA >> 3) & 1;

    // C/D layout: VGPR g, lanes 0-15: M=g; lanes 16-31: M=g+8; N=lane%16.
    const int rowD = 8 * (lane >> 4);
    const int colD = wave * 16 + (lane & 15);

    for (int t = 1; t < Tt; ++t) {
        // (1) row maxes m_b = max_j state[b,j]
        float mx = -3.402823466e38f;
#pragma unroll
        for (int q = 0; q < 8; ++q) mx = fmaxf(mx, s_state[rA][kbA + q]);
        s_pmax[rA][tid & 15] = mx;
        __syncthreads();
        if (tid < 16) {
            float m = s_pmax[tid][0];
#pragma unroll
            for (int q = 1; q < 16; ++q) m = fmaxf(m, s_pmax[tid][q]);
            s_m[tid] = m;
        }
        __syncthreads();

        // (2) P = exp(state - m), split hi/lo, store pre-swizzled (1x b128 each)
        {
            const float mrow = s_m[rA];
            v8h hv, lv;
#pragma unroll
            for (int q = 0; q < 8; ++q) {
                float p = __expf(s_state[rA][kbA + q] - mrow);
                _Float16 h = (_Float16)p;
                hv[q] = h;
                lv[q] = (_Float16)(p - (float)h);
            }
            ((v8h*)&s_Ahi[kchunkA][laneA])[halfA] = hv;
            ((v8h*)&s_Alo[kchunkA][laneA])[halfA] = lv;
        }
        __syncthreads();

        // (3) issue x_t / mask / old-state loads early (overlap with WMMA)
        float xv[8], oldv[8], mrowv[8], mskv[8];
#pragma unroll
        for (int g = 0; g < 8; ++g) {
            const int row = rowD + g;
            const int gb  = rbase + row;
            float mval = mask[(size_t)gb * Tt + t];
            mskv[g]  = mval;
            xv[g]    = y_pred[((size_t)gb * Tt + t) * Cc + colD] * mval;
            oldv[g]  = s_state[row][colD];
            mrowv[g] = s_m[row];
        }

        // (4) S = P @ E via WMMA, compensated: Ph*Eh + Ph*El + Pl*Eh
        v8f acc = {};
#pragma unroll
        for (int c = 0; c < 4; ++c) {
            v16h Ah = s_Ahi[c][lane];
            v16h Al = s_Alo[c][lane];
            acc = __builtin_amdgcn_wmma_f32_16x16x32_f16(false, Ah, false, Ehi[c],
                                                         (short)0, acc, false, false);
            acc = __builtin_amdgcn_wmma_f32_16x16x32_f16(false, Ah, false, Elo[c],
                                                         (short)0, acc, false, false);
            acc = __builtin_amdgcn_wmma_f32_16x16x32_f16(false, Al, false, Ehi[c],
                                                         (short)0, acc, false, false);
        }

        // (5) state' = m + log(S) + x_t, masked
        float res[8];
#pragma unroll
        for (int g = 0; g < 8; ++g) {
            float nw = mrowv[g] + __logf(acc[g]) + xv[g];
            res[g] = (mskv[g] > 0.f) ? nw : oldv[g];
        }
        __syncthreads();
#pragma unroll
        for (int g = 0; g < 8; ++g) s_state[rowD + g][colD] = res[g];
        __syncthreads();
    }

    // --- final logsumexp over C, subtract real path score
    float mx = -3.402823466e38f;
#pragma unroll
    for (int q = 0; q < 8; ++q) mx = fmaxf(mx, s_state[rA][kbA + q]);
    s_pmax[rA][tid & 15] = mx;
    __syncthreads();
    if (tid < 16) {
        float m = s_pmax[tid][0];
#pragma unroll
        for (int q = 1; q < 16; ++q) m = fmaxf(m, s_pmax[tid][q]);
        s_m[tid] = m;
    }
    __syncthreads();
    float sm = 0.f;
#pragma unroll
    for (int q = 0; q < 8; ++q) sm += __expf(s_state[rA][kbA + q] - s_m[rA]);
    s_pmax[rA][tid & 15] = sm;
    __syncthreads();
    if (tid < 16) {
        float s = 0.f;
#pragma unroll
        for (int q = 0; q < 16; ++q) s += s_pmax[tid][q];
        out[rbase + tid] = s_m[tid] + __logf(s) - real_in[rbase + tid];
    }
}

// ---------------------------------------------------------------------------
extern "C" void kernel_launch(void* const* d_in, const int* in_sizes, int n_in,
                              void* d_out, int out_size, void* d_ws, size_t ws_size,
                              hipStream_t stream)
{
    (void)in_sizes; (void)n_in; (void)out_size; (void)ws_size;
    const float* y_true = (const float*)d_in[0];
    const float* y_pred = (const float*)d_in[1];
    const float* mask   = (const float*)d_in[2];
    const float* trans  = (const float*)d_in[3];
    float* out  = (float*)d_out;
    float* real = (float*)d_ws;   // B floats of scratch

    crf_score_kernel<<<Bb, 256, 0, stream>>>(y_true, y_pred, mask, trans, real);
    crf_forward_kernel<<<Bb / 16, 256, 0, stream>>>(y_pred, mask, trans, real, out);
}
